// TransformerEncoderLayer_67817533604182
// MI455X (gfx1250) — matrix operand order repaired and checked by reading, not compiled, not run on hardware
//
#include <hip/hip_runtime.h>
#include <math.h>

typedef __bf16 bf16_t;
typedef __attribute__((ext_vector_type(16))) __bf16 v16bf;
typedef __attribute__((ext_vector_type(8)))  __bf16 v8bf;
typedef __attribute__((ext_vector_type(8)))  float  v8f;

#define DM    128
#define NH    8
#define HD    1024      // NH * DM
#define DFF   2048
#define SEQ   2048
#define BATCH 4
#define MTOT  8192      // BATCH * SEQ
#define EPS   1e-5f

// ---------------------------------------------------------------------------
// WMMA helpers (CDNA5 wave32, 16x16x32 bf16 -> f32)
// ---------------------------------------------------------------------------
__device__ __forceinline__ v8f vzero() {
  v8f z;
#pragma unroll
  for (int i = 0; i < 8; ++i) z[i] = 0.f;
  return z;
}

__device__ __forceinline__ v8f wmma_bf(v16bf a, v16bf b, v8f c) {
  // (neg_a, A, neg_b, B, c_mod, C, reuse_a, reuse_b)
  return __builtin_amdgcn_wmma_f32_16x16x32_bf16(false, a, false, b, (short)0, c,
                                                 false, false);
}

// A fragment: 16(M) x 32(K) tile at (row0, k0) from row-major bf16 [*, ldk].
// ISA layout: lane&15 = row; lanes 0-15 hold K 0..7 & 16..23, lanes 16-31 hold
// K 8..15 & 24..31  -> two contiguous 8-elt (16B) loads per lane.
__device__ __forceinline__ v16bf load_a_bf(const bf16_t* base, int ldk, int row0,
                                           int k0, int lane) {
  const int row = row0 + (lane & 15);
  const int kh  = (lane >> 4) << 3;
  const bf16_t* p = base + (size_t)row * ldk + (k0 + kh);
  v8bf lo = *reinterpret_cast<const v8bf*>(p);
  v8bf hi = *reinterpret_cast<const v8bf*>(p + 16);
  v16bf a;
#pragma unroll
  for (int j = 0; j < 8; ++j) { a[j] = lo[j]; a[8 + j] = hi[j]; }
  return a;
}

// Same A fragment but converting from fp32 source on the fly.
__device__ __forceinline__ v16bf load_a_f32(const float* base, int ldk, int row0,
                                            int k0, int lane) {
  const int row = row0 + (lane & 15);
  const int kh  = (lane >> 4) << 3;
  const float* p = base + (size_t)row * ldk + (k0 + kh);
  v8f lo = *reinterpret_cast<const v8f*>(p);
  v8f hi = *reinterpret_cast<const v8f*>(p + 16);
  v16bf a;
#pragma unroll
  for (int j = 0; j < 8; ++j) { a[j] = (__bf16)lo[j]; a[8 + j] = (__bf16)hi[j]; }
  return a;
}

// B fragment: 32(K) x 16(N) tile at (k0, n0) read from row-major B^T [N][ldk].
// ISA layout: lane&15 = n; lanes 0-15 hold K 0..7 & 16..23, lanes 16-31 hold
// K 8..15 & 24..31 -> two contiguous 8-elt (16B) loads per lane.
__device__ __forceinline__ v16bf load_b_bt(const bf16_t* bt, int ldk, int n0,
                                           int k0, int lane) {
  const int n = n0 + (lane & 15);
  const int k = k0 + ((lane >> 4) << 3);
  const bf16_t* p = bt + (size_t)n * ldk + k;
  v8bf lo = *reinterpret_cast<const v8bf*>(p);
  v8bf hi = *reinterpret_cast<const v8bf*>(p + 16);
  v16bf b;
#pragma unroll
  for (int j = 0; j < 8; ++j) { b[j] = lo[j]; b[8 + j] = hi[j]; }
  return b;
}

// ---------------------------------------------------------------------------
// Kernel 0: weight transpose + bf16 conversion  WT[n][k] = (bf16) W[k][n]
// ---------------------------------------------------------------------------
__global__ void wtrans_kernel(const float* __restrict__ W, bf16_t* __restrict__ WT,
                              int K, int N) {
  int idx = blockIdx.x * blockDim.x + threadIdx.x;
  if (idx < K * N) {
    int nn = idx / K;
    int kk = idx - nn * K;
    WT[idx] = (__bf16)W[(size_t)kk * N + nn];
  }
}

// ---------------------------------------------------------------------------
// Kernel 1: Q/K/V projections.  grid = (MTOT/16, 24), block = 256 (8 waves).
// y: 0-7 -> Q head y, 8-15 -> K, 16-23 -> V.  Each wave: one 16x16 tile, K=128.
// Q,K stored row-major bf16 [MTOT][HD]; V stored transposed [bh][DM][SEQ].
// ---------------------------------------------------------------------------
__global__ __launch_bounds__(256) void proj_kernel(
    const float* __restrict__ x,
    const bf16_t* __restrict__ WqT, const bf16_t* __restrict__ WkT,
    const bf16_t* __restrict__ WvT,
    bf16_t* __restrict__ Qb, bf16_t* __restrict__ Kb, bf16_t* __restrict__ Vt) {
  const int m0   = blockIdx.x * 16;
  const int y    = blockIdx.y;
  const int wave = threadIdx.x >> 5;
  const int lane = threadIdx.x & 31;
  const int which = y >> 3;          // 0=Q 1=K 2=V
  const int h     = y & 7;
  const int n0    = h * DM + wave * 16;
  const bf16_t* WT = (which == 0) ? WqT : (which == 1) ? WkT : WvT;

  v8f c = vzero();
#pragma unroll
  for (int ks = 0; ks < 4; ++ks) {
    v16bf a  = load_a_f32(x, DM, m0, ks * 32, lane);
    v16bf bb = load_b_bt(WT, DM, n0, ks * 32, lane);
    c = wmma_bf(a, bb, c);
  }
  const int rh = (lane >> 4) << 3;
  const int n  = lane & 15;
  if (which < 2) {
    bf16_t* O = (which == 0) ? Qb : Kb;
#pragma unroll
    for (int i = 0; i < 8; ++i)
      O[(size_t)(m0 + rh + i) * HD + n0 + n] = (__bf16)c[i];
  } else {
    const int d = wave * 16 + n;
#pragma unroll
    for (int i = 0; i < 8; ++i) {
      int r = m0 + rh + i;
      int b = r >> 11;            // / SEQ
      int s = r & (SEQ - 1);
      Vt[((size_t)(b * NH + h) * DM + d) * SEQ + s] = (__bf16)c[i];
    }
  }
}

// ---------------------------------------------------------------------------
// Kernel 2: flash attention.  grid = (32 bh, 128 q-blocks), block = 32 (1 wave).
// Per wave: 16 query rows, stream 2048 keys in 32-key chunks with running
// softmax; ctx accumulated in 16x128 fp32 registers.
// ---------------------------------------------------------------------------
__global__ __launch_bounds__(32) void attn_kernel(
    const bf16_t* __restrict__ Qb, const bf16_t* __restrict__ Kb,
    const bf16_t* __restrict__ Vt, bf16_t* __restrict__ Ctx) {
  __shared__ alignas(64) bf16_t pls[16][32];

  const int bh   = blockIdx.x;
  const int b    = bh >> 3, h = bh & 7;
  const int q0   = blockIdx.y * 16;
  const int lane = threadIdx.x;
  const int brow = b * SEQ;

  const bf16_t* Qbase = Qb + (size_t)h * DM;
  const bf16_t* Kbase = Kb + (size_t)brow * HD + (size_t)h * DM;
  const bf16_t* Vbase = Vt + (size_t)bh * DM * SEQ;

  // Cache the 16x128 Q tile as four A-fragments (reused for every key chunk).
  v16bf qa[4];
#pragma unroll
  for (int ks = 0; ks < 4; ++ks)
    qa[ks] = load_a_bf(Qbase, HD, brow + q0, ks * 32, lane);

  float m_run[8], l_run[8];
#pragma unroll
  for (int i = 0; i < 8; ++i) { m_run[i] = -1e30f; l_run[i] = 0.f; }
  v8f acc[8];
#pragma unroll
  for (int j = 0; j < 8; ++j) acc[j] = vzero();

  const float scale = 0.08838834764831845f;   // 1/sqrt(128)

  for (int key0 = 0; key0 < SEQ; key0 += 32) {
    // scores for 32 keys: two 16x16 tiles, K-dim 128 (4 WMMA steps each)
    v8f s0 = vzero(), s1 = vzero();
#pragma unroll
    for (int ks = 0; ks < 4; ++ks) {
      v16bf kb0 = load_b_bt(Kbase, HD, key0,      ks * 32, lane);
      v16bf kb1 = load_b_bt(Kbase, HD, key0 + 16, ks * 32, lane);
      s0 = wmma_bf(qa[ks], kb0, s0);
      s1 = wmma_bf(qa[ks], kb1, s1);
    }
    if (key0 + 32 < SEQ)
      __builtin_prefetch(Kbase + (size_t)(key0 + 32 + (lane & 15)) * HD, 0, 3);

    // streaming softmax statistics (row i+8*(lane>>4), replicated per 16-half)
    float al[8];
#pragma unroll
    for (int i = 0; i < 8; ++i) {
      float v0 = s0[i] * scale, v1 = s1[i] * scale;
      float mx = fmaxf(v0, v1);
#pragma unroll
      for (int d = 1; d < 16; d <<= 1) mx = fmaxf(mx, __shfl_xor(mx, d, 32));
      float mn = fmaxf(m_run[i], mx);
      al[i] = __expf(m_run[i] - mn);
      m_run[i] = mn;
      float p0 = __expf(v0 - mn), p1 = __expf(v1 - mn);
      s0[i] = p0; s1[i] = p1;
      float rs = p0 + p1;
#pragma unroll
      for (int d = 1; d < 16; d <<= 1) rs += __shfl_xor(rs, d, 32);
      l_run[i] = l_run[i] * al[i] + rs;
    }
#pragma unroll
    for (int j = 0; j < 8; ++j)
#pragma unroll
      for (int i = 0; i < 8; ++i) acc[j][i] *= al[i];

    // stage P (C-layout) -> LDS -> reload as A-fragment (16x32)
    __syncthreads();
    {
      const int rh = (lane >> 4) << 3, n = lane & 15;
#pragma unroll
      for (int i = 0; i < 8; ++i) {
        pls[rh + i][n]      = (__bf16)s0[i];
        pls[rh + i][16 + n] = (__bf16)s1[i];
      }
    }
    __syncthreads();
    v16bf pf = load_a_bf(&pls[0][0], 32, 0, 0, lane);

    // ctx += P @ V  (8 d-chunks of 16)
#pragma unroll
    for (int j = 0; j < 8; ++j) {
      v16bf vb = load_b_bt(Vbase, SEQ, j * 16, key0, lane);
      acc[j] = wmma_bf(pf, vb, acc[j]);
    }
  }

  // finalize: divide by l, store ctx bf16 row-major [MTOT][HD]
  {
    float inv[8];
#pragma unroll
    for (int i = 0; i < 8; ++i) inv[i] = 1.f / l_run[i];
    const int rh = (lane >> 4) << 3, n = lane & 15;
    bf16_t* crow = Ctx + (size_t)(brow + q0) * HD + (size_t)h * DM;
#pragma unroll
    for (int j = 0; j < 8; ++j)
#pragma unroll
      for (int i = 0; i < 8; ++i)
        crow[(size_t)(rh + i) * HD + j * 16 + n] = (__bf16)(acc[j][i] * inv[i]);
  }
}

// ---------------------------------------------------------------------------
// Kernel 3: sa = ctx @ Wo, residual + LayerNorm(g1) -> X1 (f32 + bf16 copy).
// grid = MTOT/16, block = 256 (8 waves; wave w covers cols 16w..16w+15, K=1024)
// ---------------------------------------------------------------------------
__global__ __launch_bounds__(256) void out_ln_kernel(
    const bf16_t* __restrict__ Ctx, const bf16_t* __restrict__ WoT,
    const float* __restrict__ x, const float* __restrict__ g1,
    float* __restrict__ X1f, bf16_t* __restrict__ X1b) {
  __shared__ alignas(32) float sa[16][DM];
  __shared__ float mu_s[16], rs_s[16];
  const int m0   = blockIdx.x * 16;
  const int wave = threadIdx.x >> 5, lane = threadIdx.x & 31;

  v8f c = vzero();
  for (int k0 = 0; k0 < HD; k0 += 32) {
    v16bf a  = load_a_bf(Ctx, HD, m0, k0, lane);
    v16bf bb = load_b_bt(WoT, HD, wave * 16, k0, lane);
    c = wmma_bf(a, bb, c);
  }
  const int rh = (lane >> 4) << 3, n = lane & 15;
#pragma unroll
  for (int i = 0; i < 8; ++i) sa[rh + i][wave * 16 + n] = c[i];
  __syncthreads();

  const int tid = threadIdx.x;
  if (tid < 16) {
    const float* xr = x + (size_t)(m0 + tid) * DM;
    float s = 0.f, s2 = 0.f;
    for (int cix = 0; cix < DM; ++cix) {
      float v = sa[tid][cix] + xr[cix];
      s += v; s2 += v * v;
    }
    float mu  = s * (1.f / DM);
    float var = s2 * (1.f / DM) - mu * mu;
    mu_s[tid] = mu;
    rs_s[tid] = rsqrtf(var + EPS);
  }
  __syncthreads();
  for (int idx = tid; idx < 16 * DM; idx += 256) {
    int r = idx >> 7, cix = idx & (DM - 1);
    float v = sa[r][cix] + x[(size_t)(m0 + r) * DM + cix];
    float o = (v - mu_s[r]) * rs_s[r] * g1[cix];
    X1f[(size_t)(m0 + r) * DM + cix] = o;
    X1b[(size_t)(m0 + r) * DM + cix] = (__bf16)o;
  }
}

// ---------------------------------------------------------------------------
// Kernel 4: FFN fused: h = relu(x1@W1) held in 64KB LDS, out = LN(x1 + h@W2).
// grid = MTOT/16, block = 256 (8 waves).
// ---------------------------------------------------------------------------
__global__ __launch_bounds__(256) void ffn_kernel(
    const float* __restrict__ X1f, const bf16_t* __restrict__ X1b,
    const bf16_t* __restrict__ W1T, const bf16_t* __restrict__ W2T,
    const float* __restrict__ g2, float* __restrict__ out) {
  __shared__ alignas(32) union FfnLds {
    bf16_t hmat[16][DFF];                                    // 64 KB
    struct { float sa[16][DM]; float mu[16]; float rs[16]; } p2;
  } lds;

  const int m0   = blockIdx.x * 16;
  const int wave = threadIdx.x >> 5, lane = threadIdx.x & 31;
  const int rh   = (lane >> 4) << 3, n = lane & 15;

  // x1 tile A-fragments (K=128), reused for all 2048 output cols
  v16bf a1[4];
#pragma unroll
  for (int ks = 0; ks < 4; ++ks) a1[ks] = load_a_bf(X1b, DM, m0, ks * 32, lane);

  // phase 1: h = relu(x1 @ W1), 128 N-tiles over 8 waves
  for (int t = 0; t < 16; ++t) {
    const int n0 = (wave + 8 * t) * 16;
    v8f c = vzero();
#pragma unroll
    for (int ks = 0; ks < 4; ++ks) {
      v16bf bb = load_b_bt(W1T, DM, n0, ks * 32, lane);
      c = wmma_bf(a1[ks], bb, c);
    }
#pragma unroll
    for (int i = 0; i < 8; ++i)
      lds.hmat[rh + i][n0 + n] = (__bf16)fmaxf(c[i], 0.f);
  }
  __syncthreads();

  // phase 2: out16 = h @ W2  (K=2048, one 16-col tile per wave)
  v8f c2 = vzero();
  const int n0 = wave * 16;
  for (int k0 = 0; k0 < DFF; k0 += 32) {
    v16bf a  = load_a_bf(&lds.hmat[0][0], DFF, 0, k0, lane);
    v16bf bb = load_b_bt(W2T, DFF, n0, k0, lane);
    c2 = wmma_bf(a, bb, c2);
  }
  __syncthreads();   // all hmat reads done before aliased sa is written
#pragma unroll
  for (int i = 0; i < 8; ++i) lds.p2.sa[rh + i][n0 + n] = c2[i];
  __syncthreads();

  const int tid = threadIdx.x;
  if (tid < 16) {
    const float* xr = X1f + (size_t)(m0 + tid) * DM;
    float s = 0.f, s2 = 0.f;
    for (int cix = 0; cix < DM; ++cix) {
      float v = lds.p2.sa[tid][cix] + xr[cix];
      s += v; s2 += v * v;
    }
    float mu  = s * (1.f / DM);
    float var = s2 * (1.f / DM) - mu * mu;
    lds.p2.mu[tid] = mu;
    lds.p2.rs[tid] = rsqrtf(var + EPS);
  }
  __syncthreads();
  for (int idx = tid; idx < 16 * DM; idx += 256) {
    int r = idx >> 7, cix = idx & (DM - 1);
    float v = lds.p2.sa[r][cix] + X1f[(size_t)(m0 + r) * DM + cix];
    float o = (v - lds.p2.mu[r]) * lds.p2.rs[r] * g2[cix];
    out[(size_t)(m0 + r) * DM + cix] = o;
  }
}

// ---------------------------------------------------------------------------
// Host launcher
// ---------------------------------------------------------------------------
extern "C" void kernel_launch(void* const* d_in, const int* in_sizes, int n_in,
                              void* d_out, int out_size, void* d_ws, size_t ws_size,
                              hipStream_t stream) {
  const float* x  = (const float*)d_in[0];
  const float* Wq = (const float*)d_in[1];
  const float* Wk = (const float*)d_in[2];
  const float* Wv = (const float*)d_in[3];
  const float* Wo = (const float*)d_in[4];
  const float* W1 = (const float*)d_in[5];
  const float* W2 = (const float*)d_in[6];
  const float* g1 = (const float*)d_in[7];
  const float* g2 = (const float*)d_in[8];
  float* out = (float*)d_out;

  char* ws = (char*)d_ws;
  size_t off = 0;
  auto take = [&](size_t bytes) -> char* {
    char* p = ws + off;
    off += (bytes + 255) & ~(size_t)255;
    return p;
  };
  bf16_t* WqT = (bf16_t*)take((size_t)HD * DM * 2);       // [1024][128]
  bf16_t* WkT = (bf16_t*)take((size_t)HD * DM * 2);
  bf16_t* WvT = (bf16_t*)take((size_t)HD * DM * 2);
  bf16_t* WoT = (bf16_t*)take((size_t)DM * HD * 2);       // [128][1024]
  bf16_t* W1T = (bf16_t*)take((size_t)DFF * DM * 2);      // [2048][128]
  bf16_t* W2T = (bf16_t*)take((size_t)DM * DFF * 2);      // [128][2048]
  bf16_t* Qb  = (bf16_t*)take((size_t)MTOT * HD * 2);     // 16 MB
  bf16_t* Kb  = (bf16_t*)take((size_t)MTOT * HD * 2);     // 16 MB
  bf16_t* Vt  = (bf16_t*)take((size_t)MTOT * HD * 2);     // 16 MB  [bh][d][s]
  bf16_t* Ctx = (bf16_t*)take((size_t)MTOT * HD * 2);     // 16 MB
  float*  X1f = (float*) take((size_t)MTOT * DM * 4);     // 4 MB
  bf16_t* X1b = (bf16_t*)take((size_t)MTOT * DM * 2);     // 2 MB
  (void)ws_size; (void)in_sizes; (void)n_in; (void)out_size;

  // weight prep
  wtrans_kernel<<<(DM * HD + 255) / 256, 256, 0, stream>>>(Wq, WqT, DM, HD);
  wtrans_kernel<<<(DM * HD + 255) / 256, 256, 0, stream>>>(Wk, WkT, DM, HD);
  wtrans_kernel<<<(DM * HD + 255) / 256, 256, 0, stream>>>(Wv, WvT, DM, HD);
  wtrans_kernel<<<(HD * DM + 255) / 256, 256, 0, stream>>>(Wo, WoT, HD, DM);
  wtrans_kernel<<<(DM * DFF + 255) / 256, 256, 0, stream>>>(W1, W1T, DM, DFF);
  wtrans_kernel<<<(DFF * DM + 255) / 256, 256, 0, stream>>>(W2, W2T, DFF, DM);

  proj_kernel<<<dim3(MTOT / 16, 24), 256, 0, stream>>>(x, WqT, WkT, WvT, Qb, Kb, Vt);
  attn_kernel<<<dim3(BATCH * NH, SEQ / 16), 32, 0, stream>>>(Qb, Kb, Vt, Ctx);
  out_ln_kernel<<<MTOT / 16, 256, 0, stream>>>(Ctx, WoT, x, g1, X1f, X1b);
  ffn_kernel<<<MTOT / 16, 256, 0, stream>>>(X1f, X1b, W1T, W2T, g2, out);
}
